// GPT2Attention_22101901705524
// MI455X (gfx1250) — compile-verified, matrix operand order checked by reading, not run on hardware
//
#include <hip/hip_runtime.h>
#include <hip/hip_bf16.h>

// ---------------------------------------------------------------------------
// GPT-2 causal self-attention forward, bf16 WMMA path for gfx1250 (MI455X).
// B=2, T=2048, C=1024, H=16, Dh=64.
// GEMMs: double-buffered LDS slabs fed by BOTH CDNA5 DMA paths:
//   A-slab: Tensor Data Mover (tensor_load_to_lds, TENSORcnt)
//   B-slab: per-lane async copies (global_load_async_to_lds_b128, ASYNCcnt)
// feeding v_wmma_f32_16x16x32_bf16. Attention: register-resident flash kernel.
// ---------------------------------------------------------------------------

typedef __bf16 bf16;
typedef __attribute__((ext_vector_type(16))) __bf16 v16bf;
typedef __attribute__((ext_vector_type(8)))  __bf16 v8bf;
typedef __attribute__((ext_vector_type(8)))  float  v8f;
typedef __attribute__((ext_vector_type(4)))  unsigned int u32x4;
typedef __attribute__((ext_vector_type(8)))  unsigned int u32x8;

#define BATCH   2
#define SEQ     2048
#define CH      1024
#define NHEAD   16
#define HDIM    64
#define NROWS   (BATCH * SEQ)     // 4096
#define NEG_BIG (-3.0e38f)
#define LDS_STRIDE 40             // 32 k-elements + 8 pad (80B rows, 16B aligned)

union FragU { v16bf f; v8bf h[2]; };

__device__ __forceinline__ v8f wmma_bf16(v16bf a, v16bf b, v8f c) {
  // (neg_a, A, neg_b, B, c_mod, C, reuse_a, reuse_b)
  return __builtin_amdgcn_wmma_f32_16x16x32_bf16(false, a, false, b,
                                                 (short)0, c, false, false);
}

// 16x32 bf16 A-fragment (or mirror 32x16 B-fragment for NT GEMMs) from
// row-major GLOBAL memory. ISA layout (CDNA5 ISA):
//   lane l (l<16):  row M = l,     K = {0..7} in f[0..7], {16..23} in f[8..15]
//   lane l (l>=16): row M = l-16,  K = {8..15},           {24..31}
__device__ __forceinline__ v16bf load_frag(const bf16* __restrict__ base,
                                           int row_stride, int row, int k) {
  int lane = threadIdx.x & 31;
  const bf16* p = base + (size_t)(row + (lane & 15)) * (size_t)row_stride
                       + (size_t)(k + ((lane >> 4) << 3));
  FragU u;
  u.h[0] = *reinterpret_cast<const v8bf*>(p);
  u.h[1] = *reinterpret_cast<const v8bf*>(p + 16);
  return u.f;
}

// Same fragment sourced from an LDS slab with LDS_STRIDE-element rows.
__device__ __forceinline__ v16bf lds_frag(const bf16* base) {
  int lane = threadIdx.x & 31;
  const bf16* p = base + (lane & 15) * LDS_STRIDE + ((lane >> 4) << 3);
  FragU u;
  u.h[0] = *reinterpret_cast<const v8bf*>(p);
  u.h[1] = *reinterpret_cast<const v8bf*>(p + 16);
  return u.f;
}

// ---- CDNA5 Tensor Data Mover: DMA a rows x 32-element bf16 2D tile into LDS.
// D# per CDNA5 ISA:
//   group0: count=1 | lds_addr[63:32] | global_addr[120:64] | type=2[127:126]
//   group1: data_size=1 (2B), pad_enable, pad_interval=3 (16 DWORDs = one 64B
//           row), pad_amount=3 (4 DWORDs = 16B) -> LDS row stride 80B, exactly
//           LDS_STRIDE.  tile_dim0=32 elems, tile_dim1=128 rows,
//           tensor_dim0_stride=CH.  Tracked on TENSORcnt.
__device__ __forceinline__ void tdm_load_slab(const bf16* gtile,
                                              unsigned lds_byte_addr,
                                              unsigned tensor_rows) {
  const unsigned long long ga = (unsigned long long)(uintptr_t)gtile;
  u32x4 g0;
  g0.x = 1u;                                   // count=1 (valid), user mode
  g0.y = lds_byte_addr;                        // lds_addr
  g0.z = (unsigned)ga;                         // global_addr[31:0]
  g0.w = (unsigned)(ga >> 32) | (2u << 30);    // global_addr[56:32] | type=2
  u32x8 g1;
  g1[0] = (1u << 16)                           // data_size = 1 -> 2 bytes
        | (1u << 20)                           // pad_enable
        | (3u << 22)                           // pad_interval: 16 DWORDs
        | (3u << 25);                          // pad_amount:   4 DWORDs
  g1[1] = (CH & 0xffffu) << 16;                // tensor_dim0[15:0] = 1024
  g1[2] = ((unsigned)CH >> 16)                 // tensor_dim0[31:16]
        | ((tensor_rows & 0xffffu) << 16);     // tensor_dim1[15:0]
  g1[3] = (tensor_rows >> 16)                  // tensor_dim1[31:16]
        | (32u << 16);                         // tile_dim0 = 32 elements
  g1[4] = 128u;                                // tile_dim1 = 128 rows (dim2=0)
  g1[5] = (unsigned)CH;                        // tensor_dim0_stride[31:0]
  g1[6] = 0u;                                  // stride0 hi / dim1_stride lo
  g1[7] = 0u;                                  // dim1_stride hi
  asm volatile("tensor_load_to_lds %0, %1" :: "s"(g0), "s"(g1) : "memory");
}

__device__ __forceinline__ void wait_tensorcnt0() {
  asm volatile("s_wait_tensorcnt 0x0" ::: "memory");
}

// ---- per-lane async-DMA of one 128x32 bf16 slab into LDS (ASYNCcnt) ----
__device__ __forceinline__ void async_tile_load(const bf16* __restrict__ gbase,
                                                int grow0, int gstride, int k,
                                                bf16* lbuf) {
  const int tid = threadIdx.x;
#pragma unroll
  for (int i = 0; i < 2; ++i) {
    const int c   = tid + i * 256;       // chunk id 0..511
    const int row = c >> 2;              // 0..127
    const int kc  = (c & 3) * 8;         // 0,8,16,24 (elements)
    const bf16* g = gbase + (size_t)(grow0 + row) * (size_t)gstride
                          + (size_t)(k + kc);
    const unsigned lds = (unsigned)(uintptr_t)(lbuf + row * LDS_STRIDE + kc);
    asm volatile("global_load_async_to_lds_b128 %0, %1, off"
                 :: "v"(lds), "v"(g) : "memory");
  }
}

__device__ __forceinline__ void wait_asynccnt0() {
  asm volatile("s_wait_asynccnt 0x0" ::: "memory");
}

// ---------------------------------------------------------------------------
// fp32 -> bf16 conversion
// ---------------------------------------------------------------------------
__global__ void cvt_f32_bf16_kernel(const float* __restrict__ src,
                                    bf16* __restrict__ dst, int n) {
  int i = blockIdx.x * blockDim.x + threadIdx.x;
  if (i < n) dst[i] = (bf16)src[i];
}

// ---------------------------------------------------------------------------
// GEMM: out[n,o] = sum_c A[n,c] * W[o,c] + bias[o]
// A: [NROWS, CH] bf16 row-major.  W: [CH, CH] bf16 row-major (rows = out chan)
// mode 0: write bf16 to [B,H,T,Dh]     (Q, K)
// mode 1: write bf16 to [B,H,Dh,T]     (V transposed for the PV matmul)
// mode 2: write fp32 to [NROWS, CH]    (final output projection)
// Block = 256 threads (8 waves, 4(M) x 2(N)), 128x128 tile.
// K-loop: A-slab by TDM (wave 0), B-slab by per-lane async; double-buffered;
// 8 WMMAs per 32-wide K-step.
// ---------------------------------------------------------------------------
__global__ __launch_bounds__(256) void gemm_bf16_kernel(
    const bf16* __restrict__ A, const bf16* __restrict__ W,
    const float* __restrict__ bias, bf16* __restrict__ outb,
    float* __restrict__ outf, int mode) {
  __shared__ __attribute__((aligned(16))) bf16 sA[2][128 * LDS_STRIDE];
  __shared__ __attribute__((aligned(16))) bf16 sB[2][128 * LDS_STRIDE];

  const int wid  = threadIdx.x >> 5;
  const int lane = threadIdx.x & 31;
  const int wm = wid & 3;         // 0..3 -> 32-row strip
  const int wn = wid >> 2;        // 0..1 -> 64-col strip
  const int mB = blockIdx.x * 128;
  const int nB = blockIdx.y * 128;
  const int m0 = mB + wm * 32;
  const int n0 = nB + wn * 64;

  v8f acc[2][4];
#pragma unroll
  for (int i = 0; i < 2; ++i)
#pragma unroll
    for (int j = 0; j < 4; ++j) acc[i][j] = (v8f)0.0f;

  // Prologue: DMA first K-slab (A via TDM from wave 0, B via async lanes).
  if (wid == 0)
    tdm_load_slab(A + (size_t)mB * CH, (unsigned)(uintptr_t)sA[0], NROWS);
  async_tile_load(W, nB, CH, 0, sB[0]);

  for (int k = 0; k < CH; k += 32) {
    const int cur = (k >> 5) & 1;
    wait_asynccnt0();            // this wave's B-slab DMAs have landed
    if (wid == 0) wait_tensorcnt0();  // TDM A-slab has landed
    __syncthreads();             // publish both slabs to all waves
    if (k + 32 < CH) {           // kick next slab's DMA; overlaps WMMAs below
      if (wid == 0)
        tdm_load_slab(A + (size_t)mB * CH + (k + 32),
                      (unsigned)(uintptr_t)sA[cur ^ 1], NROWS);
      async_tile_load(W, nB, CH, k + 32, sB[cur ^ 1]);
    }

    const bf16* la = sA[cur] + (wm * 32) * LDS_STRIDE;
    const bf16* lb = sB[cur] + (wn * 64) * LDS_STRIDE;
    v16bf a0 = lds_frag(la);
    v16bf a1 = lds_frag(la + 16 * LDS_STRIDE);
    v16bf b0 = lds_frag(lb);
    v16bf b1 = lds_frag(lb + 16 * LDS_STRIDE);
    v16bf b2 = lds_frag(lb + 32 * LDS_STRIDE);
    v16bf b3 = lds_frag(lb + 48 * LDS_STRIDE);
    acc[0][0] = wmma_bf16(a0, b0, acc[0][0]);
    acc[1][0] = wmma_bf16(a1, b0, acc[1][0]);
    acc[0][1] = wmma_bf16(a0, b1, acc[0][1]);
    acc[1][1] = wmma_bf16(a1, b1, acc[1][1]);
    acc[0][2] = wmma_bf16(a0, b2, acc[0][2]);
    acc[1][2] = wmma_bf16(a1, b2, acc[1][2]);
    acc[0][3] = wmma_bf16(a0, b3, acc[0][3]);
    acc[1][3] = wmma_bf16(a1, b3, acc[1][3]);
    __syncthreads();             // all reads of buffer `cur` done before reuse
  }

  // Epilogue. C/D layout: VGPR r, lanes 0-15 -> (M=r, N=lane),
  // lanes 16-31 -> (M=r+8, N=lane-16).
  const int half = lane >> 4;
  const int nl   = lane & 15;
#pragma unroll
  for (int i = 0; i < 2; ++i) {
#pragma unroll
    for (int j = 0; j < 4; ++j) {
      const int col = n0 + j * 16 + nl;
      const float bv = bias[col];
#pragma unroll
      for (int r = 0; r < 8; ++r) {
        const int row = m0 + i * 16 + r + half * 8;
        const float val = acc[i][j][r] + bv;
        if (mode == 2) {
          outf[(size_t)row * CH + col] = val;
        } else {
          const int b_ = row >> 11;      // row / SEQ
          const int t  = row & (SEQ - 1);
          const int h_ = col >> 6;       // col / HDIM
          const int d  = col & (HDIM - 1);
          size_t idx;
          if (mode == 0)
            idx = (((size_t)(b_ * NHEAD + h_) * SEQ) + t) * HDIM + d;
          else
            idx = (((size_t)(b_ * NHEAD + h_) * HDIM) + d) * SEQ + t;
          outb[idx] = (bf16)val;
        }
      }
    }
  }
}

// ---------------------------------------------------------------------------
// Causal flash attention.
// Q,K: [B,H,T,Dh] bf16.  Vt: [B,H,Dh,T] bf16.  ctx out: [B*T, C] bf16.
// Grid: B*H*(T/64) blocks of 128 threads; wave w owns 16 query rows.
// Per 32-key block: S = Q K^T (4 WMMAs), online softmax (shfl row reductions),
// P routed through wave-private LDS to reshape C-layout -> A-layout,
// O += P V (4 WMMAs).
// ---------------------------------------------------------------------------
__global__ __launch_bounds__(128) void attn_kernel(
    const bf16* __restrict__ Q, const bf16* __restrict__ K,
    const bf16* __restrict__ Vt, bf16* __restrict__ ctx) {
  __shared__ __attribute__((aligned(16))) bf16 pbuf[4][16][32];

  const int wid  = threadIdx.x >> 5;
  const int lane = threadIdx.x & 31;
  const int half = lane >> 4;
  const int nl   = lane & 15;

  const int bh   = blockIdx.x >> 5;       // (b*NHEAD + h), T/64 = 32 q-blocks
  const int qblk = blockIdx.x & 31;
  const int q0   = qblk * 64 + wid * 16;

  const bf16* Qb = Q  + (size_t)bh * SEQ * HDIM;
  const bf16* Kb = K  + (size_t)bh * SEQ * HDIM;
  const bf16* Vb = Vt + (size_t)bh * HDIM * SEQ;

  // Q fragments live in registers for the whole kernel (Dh=64 -> 2 frags).
  const v16bf aq0 = load_frag(Qb, HDIM, q0, 0);
  const v16bf aq1 = load_frag(Qb, HDIM, q0, 32);

  float mrow[8], lrow[8];
  v8f o[4];
#pragma unroll
  for (int r = 0; r < 8; ++r) { mrow[r] = NEG_BIG; lrow[r] = 0.0f; }
#pragma unroll
  for (int j = 0; j < 4; ++j) o[j] = (v8f)0.0f;

  const float scale = 0.125f;  // 1/sqrt(64)
  const int kend = q0 + 16;    // causal: only keys < q0+16 ever needed

  for (int kb = 0; kb < kend; kb += 32) {
    if (kb + 32 < kend) {  // stream next key block ahead (global_prefetch_b8)
      __builtin_prefetch(Kb + (size_t)(kb + 32) * HDIM, 0, 1);
      __builtin_prefetch(Vb + (size_t)(kb + 32), 0, 1);
    }
    // ---- S = Q K^T for 16 queries x 32 keys (two 16-wide tiles) ----
    v16bf bk00 = load_frag(Kb, HDIM, kb,      0);
    v16bf bk01 = load_frag(Kb, HDIM, kb,      32);
    v16bf bk10 = load_frag(Kb, HDIM, kb + 16, 0);
    v16bf bk11 = load_frag(Kb, HDIM, kb + 16, 32);
    v8f s0 = (v8f)0.0f, s1 = (v8f)0.0f;
    s0 = wmma_bf16(aq0, bk00, s0);
    s0 = wmma_bf16(aq1, bk01, s0);
    s1 = wmma_bf16(aq0, bk10, s1);
    s1 = wmma_bf16(aq1, bk11, s1);

    const int key0 = kb + nl;
    const int key1 = kb + 16 + nl;
    float p0[8], p1[8];
#pragma unroll
    for (int r = 0; r < 8; ++r) {
      const int q = q0 + r + half * 8;
      float x0 = s0[r] * scale;
      float x1 = s1[r] * scale;
      if (key0 > q) x0 = NEG_BIG;
      if (key1 > q) x1 = NEG_BIG;
      p0[r] = x0; p1[r] = x1;
    }

    // ---- online softmax: per-row max / sum across the 16 lanes of a half ----
#pragma unroll
    for (int r = 0; r < 8; ++r) {
      float t = fmaxf(p0[r], p1[r]);
      t = fmaxf(t, __shfl_xor(t, 1, 32));
      t = fmaxf(t, __shfl_xor(t, 2, 32));
      t = fmaxf(t, __shfl_xor(t, 4, 32));
      t = fmaxf(t, __shfl_xor(t, 8, 32));
      const float mn = fmaxf(mrow[r], t);
      const float alpha = expf(mrow[r] - mn);
      const float e0 = expf(p0[r] - mn);
      const float e1 = expf(p1[r] - mn);
      float rs = e0 + e1;
      rs += __shfl_xor(rs, 1, 32);
      rs += __shfl_xor(rs, 2, 32);
      rs += __shfl_xor(rs, 4, 32);
      rs += __shfl_xor(rs, 8, 32);
      lrow[r] = lrow[r] * alpha + rs;
      mrow[r] = mn;
#pragma unroll
      for (int j = 0; j < 4; ++j) o[j][r] = o[j][r] * alpha;
      p0[r] = e0; p1[r] = e1;
    }

    // ---- C-layout P -> bf16 -> LDS -> A-layout fragment (wave-private; DS
    //      ops from one wave are serviced in order, no barrier needed) ----
#pragma unroll
    for (int r = 0; r < 8; ++r) {
      const int row = r + half * 8;
      pbuf[wid][row][nl]      = (bf16)p0[r];
      pbuf[wid][row][nl + 16] = (bf16)p1[r];
    }
    FragU up;
    const bf16* pp = &pbuf[wid][nl][half * 8];
    up.h[0] = *reinterpret_cast<const v8bf*>(pp);
    up.h[1] = *reinterpret_cast<const v8bf*>(pp + 16);
    const v16bf ap = up.f;

    // ---- O += P V : V stored transposed, so B-fragments are contiguous ----
#pragma unroll
    for (int j = 0; j < 4; ++j) {
      v16bf bv = load_frag(Vb, SEQ, j * 16, kb);   // "row" = d, k = key index
      o[j] = wmma_bf16(ap, bv, o[j]);
    }
  }

  // ---- normalize and store ctx[b*T + q, h*64 + d] as bf16 ----
  const int b_ = bh >> 4;
  const int h_ = bh & 15;
#pragma unroll
  for (int r = 0; r < 8; ++r) {
    const int q = q0 + r + half * 8;
    const float inv = 1.0f / lrow[r];
#pragma unroll
    for (int j = 0; j < 4; ++j) {
      const int d = j * 16 + nl;
      ctx[((size_t)(b_ * SEQ + q)) * CH + h_ * HDIM + d] = (bf16)(o[j][r] * inv);
    }
  }
}

// ---------------------------------------------------------------------------
// Host-side orchestration (all launches on `stream`; no allocation).
// Workspace layout (bytes):
//   [0,8M)   x bf16            [8M,10M)  Wq bf16   [10M,12M) Wk bf16
//   [12M,14M) Wv bf16          [14M,16M) Wo bf16
//   [16M,24M) Q  [24M,32M) K   [32M,40M) Vt        [40M,48M) ctx
// ---------------------------------------------------------------------------
extern "C" void kernel_launch(void* const* d_in, const int* in_sizes, int n_in,
                              void* d_out, int out_size, void* d_ws, size_t ws_size,
                              hipStream_t stream) {
  (void)in_sizes; (void)n_in; (void)out_size; (void)ws_size;
  const float* x  = (const float*)d_in[0];
  const float* Wq = (const float*)d_in[1];
  const float* bq = (const float*)d_in[2];
  const float* Wk = (const float*)d_in[3];
  const float* bk = (const float*)d_in[4];
  const float* Wv = (const float*)d_in[5];
  const float* bv = (const float*)d_in[6];
  const float* Wo = (const float*)d_in[7];
  const float* bo = (const float*)d_in[8];

  char* ws = (char*)d_ws;
  const size_t MB = 1u << 20;
  bf16* xb   = (bf16*)(ws + 0 * MB);
  bf16* wqb  = (bf16*)(ws + 8 * MB);
  bf16* wkb  = (bf16*)(ws + 10 * MB);
  bf16* wvb  = (bf16*)(ws + 12 * MB);
  bf16* wob  = (bf16*)(ws + 14 * MB);
  bf16* Qb   = (bf16*)(ws + 16 * MB);
  bf16* Kb   = (bf16*)(ws + 24 * MB);
  bf16* Vtb  = (bf16*)(ws + 32 * MB);
  bf16* ctxb = (bf16*)(ws + 40 * MB);

  const int nx = NROWS * CH;      // 4,194,304
  const int nw = CH * CH;         // 1,048,576
  cvt_f32_bf16_kernel<<<(nx + 255) / 256, 256, 0, stream>>>(x,  xb,  nx);
  cvt_f32_bf16_kernel<<<(nw + 255) / 256, 256, 0, stream>>>(Wq, wqb, nw);
  cvt_f32_bf16_kernel<<<(nw + 255) / 256, 256, 0, stream>>>(Wk, wkb, nw);
  cvt_f32_bf16_kernel<<<(nw + 255) / 256, 256, 0, stream>>>(Wv, wvb, nw);
  cvt_f32_bf16_kernel<<<(nw + 255) / 256, 256, 0, stream>>>(Wo, wob, nw);

  dim3 gg(NROWS / 128, CH / 128);  // (32, 8)
  gemm_bf16_kernel<<<gg, 256, 0, stream>>>(xb, wqb, bq, Qb,  nullptr, 0);
  gemm_bf16_kernel<<<gg, 256, 0, stream>>>(xb, wkb, bk, Kb,  nullptr, 0);
  gemm_bf16_kernel<<<gg, 256, 0, stream>>>(xb, wvb, bv, Vtb, nullptr, 1);

  attn_kernel<<<BATCH * NHEAD * (SEQ / 64), 128, 0, stream>>>(Qb, Kb, Vtb, ctxb);

  gemm_bf16_kernel<<<gg, 256, 0, stream>>>(ctxb, wob, bo, nullptr,
                                           (float*)d_out, 2);
}
